// BaseModel_14499809591724
// MI455X (gfx1250) — matrix operand order repaired and checked by reading, not compiled, not run on hardware
//
#include <hip/hip_runtime.h>
#include <hip/hip_bf16.h>
#include <stdint.h>

// ---------------- problem constants (match reference) ----------------
#define N_NODES   50000
#define N_EDGES   800000
#define FEAT      128
#define HID       128
#define N_GRAPHS  512
#define N_CLASSES 10
#define N_BLOCKS  3
#define N_INNER   2
#define ZCOLS     (N_BLOCKS * HID)   // 384
#define NODE_F    (N_NODES * HID)    // 6,400,000

typedef __attribute__((ext_vector_type(16))) __bf16 v16bf;
typedef __attribute__((ext_vector_type(8)))  float  v8f;

// =====================================================================
// WMMA GEMM: out[M x 128] = act(A[M x K]) * W[K x 128] (+ bias)
// A supplied as one (K=128) or two (K=256, concatenated) row-major
// 128-wide fp32 buffers (a0 for k<128, a1 for k>=128).
// One wave -> one 16x16 tile; 8 waves/block cover the N=128 dimension
// for one 16-row stripe. W is staged per 128-K chunk into LDS,
// transposed and pre-converted to bf16 (padded rows to avoid bank
// conflicts), so B-fragment reads are two contiguous 16B LDS loads.
// Emits v_wmma_f32_16x16x32_bf16 with fp32 accumulation.
// =====================================================================
template <int K>
__global__ __launch_bounds__(256)
void gemm_bf16_wmma(const float* __restrict__ a0, const float* __restrict__ a1,
                    const float* __restrict__ W,  const float* __restrict__ bias,
                    float* __restrict__ out, int relu_in) {
  constexpr int KP = 128 + 8;                 // padded LDS row (bf16 elems)
  __shared__ __bf16 wt[128 * KP];             // 34,816 bytes

  const int tid  = threadIdx.x;
  const int lane = tid & 31;
  const int wave = tid >> 5;                  // 0..7 => N tile
  const int m0   = blockIdx.x * 16;           // grid is exactly M/16
  const int n0   = wave * 16;
  const int r    = lane & 15;
  const bool hi  = lane >= 16;

  v8f acc = {};

#pragma unroll
  for (int kc = 0; kc < K; kc += 128) {
    // ---- stage this 128-K chunk of W into LDS (transposed, bf16) ----
    __syncthreads();                          // protect prior-iter readers
    const float* Wc = W + (size_t)kc * 128;   // W[k][n], row-major, n-width 128
    for (int i = tid; i < 128 * 128; i += 256) {
      int k = i >> 7, n = i & 127;
      wt[n * KP + k] = (__bf16)Wc[i];         // native cvt, transposed store
    }
    __syncthreads();

    const float* abuf = (kc == 0) ? a0 : a1;
    const float* arow = abuf + (size_t)(m0 + r) * 128;

#pragma unroll
    for (int kb = 0; kb < 128; kb += 32) {
      // ---- A fragment: 16x32 bf16, ISA layout ----
      // lanes 0-15: row=lane, elems 0..7 -> K=kb..kb+7,  8..15 -> kb+16..kb+23
      // lanes16-31: row=lane-16, elems 0..7 -> kb+8..15,  8..15 -> kb+24..31
      const int ka = kb + (hi ? 8 : 0);
      v16bf afrag;
#pragma unroll
      for (int j = 0; j < 8; ++j) {
        float x0 = arow[ka + j];
        float x1 = arow[ka + 16 + j];
        if (relu_in) { x0 = fmaxf(x0, 0.0f); x1 = fmaxf(x1, 0.0f); }
        afrag[j]     = (__bf16)x0;
        afrag[8 + j] = (__bf16)x1;
      }
      // ---- B fragment: 32x16 bf16, ISA layout ----
      // lanes 0-15: col=lane, elems 0..7 -> K=kb..kb+7, 8..15 -> kb+16..kb+23; lanes 16-31: +8
      const __bf16* bp = &wt[(n0 + r) * KP + kb + (hi ? 8 : 0)];
      v16bf bfrag;
#pragma unroll
      for (int j = 0; j < 8; ++j) { bfrag[j] = bp[j]; bfrag[8 + j] = bp[16 + j]; }   // 2x 16B ds loads

      acc = __builtin_amdgcn_wmma_f32_16x16x32_bf16(
          /*neg_a=*/false, afrag, /*neg_b=*/false, bfrag,
          /*c_mod=*/(short)0, acc, /*reuse_a=*/false, /*reuse_b=*/false);
    }
  }

  // C/D layout: VGPR v -> row (hi?8:0)+v, col = lane%16
  const int   mbase = m0 + (hi ? 8 : 0);
  const float badd  = bias ? bias[n0 + r] : 0.0f;
#pragma unroll
  for (int v = 0; v < 8; ++v) {
    out[(size_t)(mbase + v) * 128 + (n0 + r)] = acc[v] + badd;
  }
}

// =====================================================================
// Small helper kernels
// =====================================================================
__global__ void zero_f32(float* __restrict__ p, int n) {
  int i = blockIdx.x * blockDim.x + threadIdx.x;
  if (i < n) p[i] = 0.0f;
}

__global__ void deg_accum(const float* __restrict__ ew, const int* __restrict__ dst,
                          float* __restrict__ deg, int E) {
  int e = blockIdx.x * blockDim.x + threadIdx.x;
  if (e < E) atomicAdd(&deg[dst[e]], ew[e]);
}

__global__ void deg_finalize(float* __restrict__ deg, int n) {
  int i = blockIdx.x * blockDim.x + threadIdx.x;
  if (i < n) deg[i] = rsqrtf(deg[i] + 1.0f);   // store dinv in place
}

// out[n,f] = bias[f] + xw[n,f] * dinv[n]^2   (self-loop term)
__global__ void conv_self(const float* __restrict__ xw, const float* __restrict__ dinv,
                          const float* __restrict__ b, float* __restrict__ out, int n) {
  int i = blockIdx.x * blockDim.x + threadIdx.x;
  if (i < n) {
    int node = i >> 7, f = i & 127;
    float d = dinv[node];
    out[i] = b[f] + xw[i] * d * d;
  }
}

// 32 lanes per edge, 4 features per lane: out[dst] += xw[src] * norm
__global__ void edge_scatter(const float* __restrict__ xw, const float* __restrict__ dinv,
                             const float* __restrict__ ew,
                             const int* __restrict__ src, const int* __restrict__ dst,
                             float* __restrict__ out, int E) {
  int g = blockIdx.x * blockDim.x + threadIdx.x;
  int e = g >> 5;
  if (e >= E) return;
  int part = (g & 31) << 2;
  int s = src[e], d = dst[e];
  float norm = dinv[s] * ew[e] * dinv[d];
  const float4 v = *(const float4*)(xw + (size_t)s * 128 + part);
  float* o = out + (size_t)d * 128 + part;
  atomicAdd(o + 0, v.x * norm);
  atomicAdd(o + 1, v.y * norm);
  atomicAdd(o + 2, v.z * norm);
  atomicAdd(o + 3, v.w * norm);
}

// pooled[batch[n], coloff+f] += relu(jkz[n,f])
__global__ void pool_add(const float* __restrict__ jkz, const int* __restrict__ batch,
                         float* __restrict__ z, int n, int coloff) {
  int i = blockIdx.x * blockDim.x + threadIdx.x;
  if (i < n) {
    int node = i >> 7, f = i & 127;
    float v = fmaxf(jkz[i], 0.0f);
    atomicAdd(&z[(size_t)batch[node] * ZCOLS + coloff + f], v);
  }
}

// Fused head: BN (eval) -> lin1+ReLU -> lin2 -> softmax. One block per graph.
__global__ __launch_bounds__(128)
void head_kernel(const float* __restrict__ z,
                 const float* __restrict__ gamma, const float* __restrict__ beta,
                 const float* __restrict__ mean,  const float* __restrict__ var,
                 const float* __restrict__ w1, const float* __restrict__ b1,
                 const float* __restrict__ w2, const float* __restrict__ b2,
                 float* __restrict__ out) {
  __shared__ float sz[ZCOLS];
  __shared__ float sy[HID];
  __shared__ float sl[N_CLASSES];
  const int g = blockIdx.x, t = threadIdx.x;
  for (int j = t; j < ZCOLS; j += 128) {
    float v = z[(size_t)g * ZCOLS + j];
    v = (v - mean[j]) * rsqrtf(var[j] + 1e-5f) * gamma[j] + beta[j];
    sz[j] = v;
  }
  __syncthreads();
  float acc = b1[t];
  for (int k = 0; k < ZCOLS; ++k) acc += sz[k] * w1[k * 128 + t];
  sy[t] = fmaxf(acc, 0.0f);
  __syncthreads();
  if (t < N_CLASSES) {
    float a = b2[t];
    for (int k = 0; k < HID; ++k) a += sy[k] * w2[k * N_CLASSES + t];
    sl[t] = a;
  }
  __syncthreads();
  if (t == 0) {
    float mx = sl[0];
    for (int c = 1; c < N_CLASSES; ++c) mx = fmaxf(mx, sl[c]);
    float e[N_CLASSES]; float ssum = 0.0f;
    for (int c = 0; c < N_CLASSES; ++c) { e[c] = __expf(sl[c] - mx); ssum += e[c]; }
    float inv = 1.0f / ssum;
    for (int c = 0; c < N_CLASSES; ++c) out[(size_t)g * N_CLASSES + c] = e[c] * inv;
  }
}

// =====================================================================
// Launch
// =====================================================================
extern "C" void kernel_launch(void* const* d_in, const int* in_sizes, int n_in,
                              void* d_out, int out_size, void* d_ws, size_t ws_size,
                              hipStream_t stream) {
  (void)in_sizes; (void)n_in; (void)out_size; (void)ws_size;
  const float* x        = (const float*)d_in[0];
  const float* ea       = (const float*)d_in[1];
  const float* conv_w   = (const float*)d_in[2];   // [3,2,128,128]
  const float* conv_b   = (const float*)d_in[3];   // [3,2,128]
  const float* jk_w     = (const float*)d_in[4];   // [3,256,128]
  const float* jk_b     = (const float*)d_in[5];   // [3,128]
  const float* bn_gamma = (const float*)d_in[6];
  const float* bn_beta  = (const float*)d_in[7];
  const float* bn_mean  = (const float*)d_in[8];
  const float* bn_var   = (const float*)d_in[9];
  const float* lin1_w   = (const float*)d_in[10];
  const float* lin1_b   = (const float*)d_in[11];
  const float* lin2_w   = (const float*)d_in[12];
  const float* lin2_b   = (const float*)d_in[13];
  const int*   eidx     = (const int*)d_in[14];    // [2,E]
  const int*   batch    = (const int*)d_in[15];
  const int* src = eidx;
  const int* dst = eidx + N_EDGES;
  float* out = (float*)d_out;

  // workspace layout (floats)
  float* ws   = (float*)d_ws;
  float* dinv = ws;                    // 50000 (padded to 50016)
  float* xw   = ws + 50016;            // NODE_F
  float* h1   = xw + NODE_F;
  float* h2   = h1 + NODE_F;
  float* jkz  = h2 + NODE_F;
  float* zbuf = jkz + NODE_F;          // N_GRAPHS * ZCOLS

  const int TPB = 256;
  const int nf  = NODE_F;

  // zero accumulators (every call: graph-replay safe)
  zero_f32<<<(N_NODES + TPB - 1) / TPB, TPB, 0, stream>>>(dinv, N_NODES);
  zero_f32<<<(N_GRAPHS * ZCOLS + TPB - 1) / TPB, TPB, 0, stream>>>(zbuf, N_GRAPHS * ZCOLS);

  // degrees -> dinv
  deg_accum<<<(N_EDGES + TPB - 1) / TPB, TPB, 0, stream>>>(ea, dst, dinv, N_EDGES);
  deg_finalize<<<(N_NODES + TPB - 1) / TPB, TPB, 0, stream>>>(dinv, N_NODES);

  const int gemm_grid = N_NODES / 16;          // 3125 (exact: no tail)
  const int esc_grid  = (N_EDGES * 32 + TPB - 1) / TPB;
  const int nf_grid   = (nf + TPB - 1) / TPB;

  for (int l = 0; l < N_BLOCKS; ++l) {
    for (int k = 0; k < N_INNER; ++k) {
      const float* ain  = (k == 1) ? h1 : ((l == 0) ? x : jkz);
      const int    rin  = (l == 0 && k == 0) ? 0 : 1;     // raw x is not rectified
      const float* W    = conv_w + (size_t)(l * N_INNER + k) * HID * HID;
      const float* b    = conv_b + (size_t)(l * N_INNER + k) * HID;
      float*       agg  = (k == 0) ? h1 : h2;
      gemm_bf16_wmma<128><<<gemm_grid, 256, 0, stream>>>(ain, nullptr, W, nullptr,
                                                         xw, rin);
      conv_self<<<nf_grid, TPB, 0, stream>>>(xw, dinv, b, agg, nf);
      edge_scatter<<<esc_grid, TPB, 0, stream>>>(xw, dinv, ea, src, dst, agg, N_EDGES);
    }
    // JumpingKnowledge: relu(cat(relu-h1, relu-h2) @ jk_w + jk_b); relu folded at consumers
    gemm_bf16_wmma<256><<<gemm_grid, 256, 0, stream>>>(
        h1, h2, jk_w + (size_t)l * 256 * HID, jk_b + (size_t)l * HID,
        jkz, 1);
    pool_add<<<nf_grid, TPB, 0, stream>>>(jkz, batch, zbuf, nf, l * HID);
  }

  head_kernel<<<N_GRAPHS, 128, 0, stream>>>(zbuf, bn_gamma, bn_beta, bn_mean, bn_var,
                                            lin1_w, lin1_b, lin2_w, lin2_b, out);
}
